// LanguageModel_89472758710273
// MI455X (gfx1250) — matrix operand order repaired and checked by reading, compiled, not run on hardware
//
#include <hip/hip_runtime.h>
#include <hip/hip_bf16.h>
#include <math.h>

// Problem constants (from reference)
#define Bb    64
#define Tt    120
#define TITLE 20
#define Hh    768
#define Ll    12
#define NHh   12
#define DHh   64
#define FFf   3072
#define Vv    21128
#define MTOK  (Bb * Tt)        // 7680 tokens
#define VC    2688             // vocab chunk (multiple of 64; tail 2312 is mult of 8)
#define LNEPS 1e-12f

// Workspace requirement: 6*MTOK*H + MTOK*FF + 3*MTOK floats ~= 236 MB.

typedef __attribute__((ext_vector_type(16))) _Float16 v16h;
typedef __attribute__((ext_vector_type(8)))  _Float16 v8h;
typedef __attribute__((ext_vector_type(2)))  _Float16 v2h;
typedef __attribute__((ext_vector_type(8)))  float    v8f;

// ---------------------------------------------------------------------------
// WMMA GEMM: C[M,N] = epi(A[M,K] * B[K,N] + bias)   (A,B fp32 -> f16, acc f32)
// Block: 256 threads = 8 waves. Tile: BM=64 x BN=64, BK=32, double-buffered.
// Wave (wave>>1) -> M subtile 0..3; (wave&1) -> 32-wide N strip; each wave
// runs TWO wmma per panel sharing one A fragment.
// Requires: M % 64 == 0, K % 32 == 0, N % 8 == 0 (clamped tail loads).
// EPI: 0 = bias, 1 = bias + exact GELU
// ---------------------------------------------------------------------------
#define BM 64
#define BN 64
#define BK 32
#define LPAD 8   // f16 pad -> row stride 40 f16 = 80 B (16B aligned rows)

template <int EPI>
__global__ __launch_bounds__(256) void gemm_wmma(
    const float* __restrict__ A, int lda,
    const float* __restrict__ Bw, int ldb,
    const float* __restrict__ bias,
    float* __restrict__ C, int ldc,
    int M, int N, int K)
{
  __shared__ __align__(16) _Float16 As[2][BM][BK + LPAD];
  __shared__ __align__(16) _Float16 Bst[2][BN][BK + LPAD];  // transposed: [n][k]

  const int tid  = threadIdx.x;
  const int lane = tid & 31;
  const int wave = tid >> 5;
  const int wm   = (wave >> 1) * 16;   // 0,16,32,48
  const int wn   = (wave & 1) * 32;    // 0,32
  const int m0   = blockIdx.y * BM;
  const int n0   = blockIdx.x * BN;

  // A staging: 64x32 floats, 8 per thread (two float4 along K)
  const int aRow = (tid * 8) >> 5, aCol = (tid * 8) & 31;
  // B staging: 32x64 floats; each thread owns a 2(K) x 4(N) sub-block so the
  // two K values per column pack into a single dword LDS store (transposed).
  const int bKr = (tid >> 4) * 2;      // 0,2,...,30
  const int bNc = (tid & 15) * 4;      // 0,4,...,60
  // Clamped, always-aligned source column (N % 4 == 0 => gn < N implies
  // gn+3 < N; clamped lanes feed only columns >= N which are never stored).
  const int gnB  = n0 + bNc;
  const int gnSafe = (gnB + 4 <= N) ? gnB : (N - 4);

  v8f acc0 = {};
  v8f acc1 = {};

  auto stage = [&](int k0, int p) {
    const float* srcA = A + (size_t)(m0 + aRow) * lda + k0 + aCol;
    float4 fa0 = *(const float4*)(srcA);
    float4 fa1 = *(const float4*)(srcA + 4);
    _Float16* dA = &As[p][aRow][aCol];
    dA[0] = (_Float16)fa0.x; dA[1] = (_Float16)fa0.y;
    dA[2] = (_Float16)fa0.z; dA[3] = (_Float16)fa0.w;
    dA[4] = (_Float16)fa1.x; dA[5] = (_Float16)fa1.y;
    dA[6] = (_Float16)fa1.z; dA[7] = (_Float16)fa1.w;

    const float* srcB0 = Bw + (size_t)(k0 + bKr) * ldb + gnSafe;
    const float* srcB1 = srcB0 + ldb;
    float4 fb0 = *(const float4*)(srcB0);   // K row bKr,   cols n..n+3
    float4 fb1 = *(const float4*)(srcB1);   // K row bKr+1, cols n..n+3
    v2h pk;
    pk[0] = (_Float16)fb0.x; pk[1] = (_Float16)fb1.x;
    *(v2h*)&Bst[p][bNc + 0][bKr] = pk;
    pk[0] = (_Float16)fb0.y; pk[1] = (_Float16)fb1.y;
    *(v2h*)&Bst[p][bNc + 1][bKr] = pk;
    pk[0] = (_Float16)fb0.z; pk[1] = (_Float16)fb1.z;
    *(v2h*)&Bst[p][bNc + 2][bKr] = pk;
    pk[0] = (_Float16)fb0.w; pk[1] = (_Float16)fb1.w;
    *(v2h*)&Bst[p][bNc + 3][bKr] = pk;

    if (k0 + BK < K) {   // prefetch the panel after this one
      __builtin_prefetch(srcA + BK, 0, 3);
      __builtin_prefetch(srcB0 + (size_t)BK * ldb, 0, 3);
    }
  };

  stage(0, 0);
  __syncthreads();

  int p = 0;
  for (int k0 = 0; k0 < K; k0 += BK) {
    if (k0 + BK < K) stage(k0 + BK, p ^ 1);   // overlap next-panel staging

    // ---- A fragment (K contiguous, matching B) ----
    // lanes 0-15: M=lane,    K 0..15
    // lanes16-31: M=lane-16, K 16..31
    const int mrow = wm + (lane & 15);
    const int koff = (lane >> 4) * 16;
    v8h a_lo = *(const v8h*)&As[p][mrow][koff];
    v8h a_hi = *(const v8h*)&As[p][mrow][koff + 8];
    v16h a;
    #pragma unroll
    for (int i = 0; i < 8; ++i) { a[i] = a_lo[i]; a[8 + i] = a_hi[i]; }

    // ---- B fragments: lane owns column n; K contiguous 0..15 / 16..31 ----
    const int kb = (lane >> 4) * 16;
    {
      const int nrow = wn + (lane & 15);
      v8h b_lo = *(const v8h*)&Bst[p][nrow][kb];
      v8h b_hi = *(const v8h*)&Bst[p][nrow][kb + 8];
      v16h b;
      #pragma unroll
      for (int i = 0; i < 8; ++i) { b[i] = b_lo[i]; b[8 + i] = b_hi[i]; }
      acc0 = __builtin_amdgcn_wmma_f32_16x16x32_f16(
          false, a, false, b, (short)0, acc0, false, false);
    }
    {
      const int nrow = wn + 16 + (lane & 15);
      v8h b_lo = *(const v8h*)&Bst[p][nrow][kb];
      v8h b_hi = *(const v8h*)&Bst[p][nrow][kb + 8];
      v16h b;
      #pragma unroll
      for (int i = 0; i < 8; ++i) { b[i] = b_lo[i]; b[8 + i] = b_hi[i]; }
      acc1 = __builtin_amdgcn_wmma_f32_16x16x32_f16(
          false, a, false, b, (short)0, acc1, false, false);
    }

    __syncthreads();
    p ^= 1;
  }

  // ---- epilogue: C/D layout — VGPR r: M = r + (lane>=16 ? 8:0), N = lane&15
  const int lm = (lane >> 4) * 8;
  {
    const int cn = n0 + wn + (lane & 15);
    if (cn < N) {
      const float bv = bias ? bias[cn] : 0.f;
      #pragma unroll
      for (int r = 0; r < 8; ++r) {
        float val = acc0[r] + bv;
        if (EPI == 1)
          val = 0.5f * val * (1.0f + erff(val * 0.70710678118654752440f));
        C[(size_t)(m0 + wm + r + lm) * ldc + cn] = val;
      }
    }
  }
  {
    const int cn = n0 + wn + 16 + (lane & 15);
    if (cn < N) {
      const float bv = bias ? bias[cn] : 0.f;
      #pragma unroll
      for (int r = 0; r < 8; ++r) {
        float val = acc1[r] + bv;
        if (EPI == 1)
          val = 0.5f * val * (1.0f + erff(val * 0.70710678118654752440f));
        C[(size_t)(m0 + wm + r + lm) * ldc + cn] = val;
      }
    }
  }
}

// ---------------------------------------------------------------------------
// Embedding sum: word + pos + type(0)  -> out[token][H]
// ---------------------------------------------------------------------------
__global__ __launch_bounds__(256) void embed_kernel(
    const int* __restrict__ x, const float* __restrict__ we,
    const float* __restrict__ pe, const float* __restrict__ te,
    float* __restrict__ out)
{
  const int token = blockIdx.x;
  const int t = token % Tt;
  const int id = x[token];
  const float* w = we + (size_t)id * Hh;
  const float* p = pe + (size_t)t * Hh;
  float* o = out + (size_t)token * Hh;
  for (int i = threadIdx.x; i < Hh; i += 256) o[i] = w[i] + p[i] + te[i];
}

// ---------------------------------------------------------------------------
// out = LN(resid + xin) * g + b   (resid nullable; in-place on resid is safe)
// ---------------------------------------------------------------------------
__global__ __launch_bounds__(256) void add_ln_kernel(
    const float* __restrict__ resid, const float* __restrict__ xin,
    const float* __restrict__ g, const float* __restrict__ bta,
    float* __restrict__ out)
{
  __shared__ float buf[Hh];
  __shared__ float reds[256], redq[256];
  __shared__ float s_mean, s_inv;
  const int token = blockIdx.x;
  const float* xr = xin + (size_t)token * Hh;
  const float* rr = resid ? resid + (size_t)token * Hh : nullptr;
  float ls = 0.f, lq = 0.f;
  for (int i = threadIdx.x; i < Hh; i += 256) {
    float v = xr[i] + (rr ? rr[i] : 0.f);
    buf[i] = v; ls += v; lq += v * v;
  }
  reds[threadIdx.x] = ls; redq[threadIdx.x] = lq;
  __syncthreads();
  for (int off = 128; off > 0; off >>= 1) {
    if (threadIdx.x < off) {
      reds[threadIdx.x] += reds[threadIdx.x + off];
      redq[threadIdx.x] += redq[threadIdx.x + off];
    }
    __syncthreads();
  }
  if (threadIdx.x == 0) {
    float mean = reds[0] * (1.0f / Hh);
    float var  = redq[0] * (1.0f / Hh) - mean * mean;
    s_mean = mean; s_inv = rsqrtf(var + LNEPS);
  }
  __syncthreads();
  float* o = out + (size_t)token * Hh;
  for (int i = threadIdx.x; i < Hh; i += 256)
    o[i] = (buf[i] - s_mean) * s_inv * g[i] + bta[i];
}

// ---------------------------------------------------------------------------
// Fused attention: one wave per (b, head, q-row). UniLM mask inline,
// shuffle softmax, ctx matvec via LDS probabilities.
// ---------------------------------------------------------------------------
__global__ __launch_bounds__(256) void attn_kernel(
    const float* __restrict__ q, const float* __restrict__ k,
    const float* __restrict__ v, float* __restrict__ ctx)
{
  __shared__ float pbuf[8][Tt];
  const int wave = threadIdx.x >> 5;
  const int lane = threadIdx.x & 31;
  const int gw   = blockIdx.x * 8 + wave;
  const int t    = gw % Tt;
  const int rem  = gw / Tt;
  const int head = rem % NHh;
  const int b    = rem / NHh;
  const float* qrow = q + ((size_t)(b * Tt + t) * Hh + head * DHh);
  const float scale = 0.125f;   // 1/sqrt(64)

  float sv[4];
  float m = -3.0e38f;
  #pragma unroll
  for (int i = 0; i < 4; ++i) {
    const int kk = lane + i * 32;
    float s = -3.0e38f;
    if (kk < Tt) {
      const float* krow = k + ((size_t)(b * Tt + kk) * Hh + head * DHh);
      float acc = 0.f;
      for (int d = 0; d < DHh; ++d) acc += qrow[d] * krow[d];
      const bool allowed = (kk < TITLE) || (t >= TITLE && kk <= t);
      s = acc * scale + (allowed ? 0.f : -10000.f);
    }
    sv[i] = s; m = fmaxf(m, s);
  }
  for (int off = 16; off > 0; off >>= 1) m = fmaxf(m, __shfl_xor(m, off, 32));
  float lsum = 0.f;
  #pragma unroll
  for (int i = 0; i < 4; ++i) {
    const int kk = lane + i * 32;
    float e = (kk < Tt) ? expf(sv[i] - m) : 0.f;
    sv[i] = e; lsum += e;
  }
  for (int off = 16; off > 0; off >>= 1) lsum += __shfl_xor(lsum, off, 32);
  const float inv = 1.0f / lsum;
  #pragma unroll
  for (int i = 0; i < 4; ++i) {
    const int kk = lane + i * 32;
    if (kk < Tt) pbuf[wave][kk] = sv[i] * inv;
  }
  __syncthreads();
  float* orow = ctx + ((size_t)(b * Tt + t) * Hh + head * DHh);
  #pragma unroll
  for (int j = 0; j < 2; ++j) {
    const int d = lane * 2 + j;
    float acc = 0.f;
    for (int kk = 0; kk < Tt; ++kk)
      acc += pbuf[wave][kk] * v[(size_t)(b * Tt + kk) * Hh + head * DHh + d];
    orow[d] = acc;
  }
}

// ---------------------------------------------------------------------------
// Streaming log-sum-exp over vocab chunks
// ---------------------------------------------------------------------------
__global__ void loss_init_kernel(float* rmax, float* rsum, float* tlog)
{
  const int i = blockIdx.x * 256 + threadIdx.x;
  if (i < MTOK) { rmax[i] = -3.0e38f; rsum[i] = 0.f; tlog[i] = 0.f; }
}

__global__ __launch_bounds__(256) void loss_chunk_kernel(
    const float* __restrict__ Lc, const int* __restrict__ y,
    int n0, int n, float* rmax, float* rsum, float* tlog)
{
  __shared__ float red[256];
  const int token = blockIdx.x;
  const float* row = Lc + (size_t)token * VC;
  float lmax = -3.0e38f;
  for (int c = threadIdx.x; c < n; c += 256) lmax = fmaxf(lmax, row[c]);
  red[threadIdx.x] = lmax; __syncthreads();
  for (int off = 128; off > 0; off >>= 1) {
    if (threadIdx.x < off)
      red[threadIdx.x] = fmaxf(red[threadIdx.x], red[threadIdx.x + off]);
    __syncthreads();
  }
  const float newmax = fmaxf(red[0], rmax[token]);
  __syncthreads();
  float ls = 0.f;
  for (int c = threadIdx.x; c < n; c += 256) ls += expf(row[c] - newmax);
  red[threadIdx.x] = ls; __syncthreads();
  for (int off = 128; off > 0; off >>= 1) {
    if (threadIdx.x < off) red[threadIdx.x] += red[threadIdx.x + off];
    __syncthreads();
  }
  if (threadIdx.x == 0) {
    const float om = rmax[token];
    rsum[token] = rsum[token] * expf(om - newmax) + red[0];
    rmax[token] = newmax;
    const int yv = y[token];
    if (yv >= n0 && yv < n0 + n) tlog[token] = row[yv - n0];
  }
}

__global__ __launch_bounds__(256) void loss_final_kernel(
    const float* rmax, const float* rsum, const float* tlog, float* out)
{
  __shared__ float red[256];
  float s = 0.f;
  for (int i = threadIdx.x; i < MTOK; i += 256)
    s += rmax[i] + logf(rsum[i]) - tlog[i];
  red[threadIdx.x] = s; __syncthreads();
  for (int off = 128; off > 0; off >>= 1) {
    if (threadIdx.x < off) red[threadIdx.x] += red[threadIdx.x + off];
    __syncthreads();
  }
  if (threadIdx.x == 0) out[0] = red[0] / (float)MTOK;
}

// ---------------------------------------------------------------------------
extern "C" void kernel_launch(void* const* d_in, const int* in_sizes, int n_in,
                              void* d_out, int out_size, void* d_ws, size_t ws_size,
                              hipStream_t stream)
{
  const int*   x    = (const int*)  d_in[0];
  const int*   y    = (const int*)  d_in[1];
  const float* we   = (const float*)d_in[2];
  const float* pe   = (const float*)d_in[3];
  const float* te   = (const float*)d_in[4];
  const float* elg  = (const float*)d_in[5];
  const float* elb  = (const float*)d_in[6];
  const float* Wq   = (const float*)d_in[7];
  const float* Wk   = (const float*)d_in[8];
  const float* Wv   = (const float*)d_in[9];
  const float* Wo   = (const float*)d_in[10];
  const float* W1   = (const float*)d_in[11];
  const float* W2   = (const float*)d_in[12];
  const float* bq   = (const float*)d_in[13];
  const float* bk   = (const float*)d_in[14];
  const float* bv   = (const float*)d_in[15];
  const float* bo   = (const float*)d_in[16];
  const float* b1   = (const float*)d_in[17];
  const float* b2   = (const float*)d_in[18];
  const float* ln1g = (const float*)d_in[19];
  const float* ln1b = (const float*)d_in[20];
  const float* ln2g = (const float*)d_in[21];
  const float* ln2b = (const float*)d_in[22];
  const float* Wc   = (const float*)d_in[23];
  const float* bc   = (const float*)d_in[24];

  float* ws = (float*)d_ws;
  const size_t MH = (size_t)MTOK * Hh;
  float* h    = ws;
  float* qb   = h  + MH;
  float* kb   = qb + MH;
  float* vb   = kb + MH;
  float* cb   = vb + MH;
  float* tb   = cb + MH;
  float* a1   = tb + MH;                       // MTOK*FF; also reused as Lc
  float* Lc   = a1;                            // MTOK*VC <= MTOK*FF
  float* rmax = a1 + (size_t)MTOK * FFf;
  float* rsum = rmax + MTOK;
  float* tlg  = rsum + MTOK;

  const dim3 blk(256);
  const dim3 gH(Hh / BN, MTOK / BM);           // (12, 120)
  const dim3 gFF(FFf / BN, MTOK / BM);         // (48, 120)

  // Embedding + LN
  embed_kernel<<<MTOK, blk, 0, stream>>>(x, we, pe, te, tb);
  add_ln_kernel<<<MTOK, blk, 0, stream>>>(nullptr, tb, elg, elb, h);

  for (int l = 0; l < Ll; ++l) {
    const float* wq = Wq + (size_t)l * Hh * Hh;
    const float* wk = Wk + (size_t)l * Hh * Hh;
    const float* wv = Wv + (size_t)l * Hh * Hh;
    const float* wo = Wo + (size_t)l * Hh * Hh;
    const float* w1 = W1 + (size_t)l * Hh * FFf;
    const float* w2 = W2 + (size_t)l * FFf * Hh;

    gemm_wmma<0><<<gH, blk, 0, stream>>>(h, Hh, wq, Hh, bq + (size_t)l * Hh,
                                         qb, Hh, MTOK, Hh, Hh);
    gemm_wmma<0><<<gH, blk, 0, stream>>>(h, Hh, wk, Hh, bk + (size_t)l * Hh,
                                         kb, Hh, MTOK, Hh, Hh);
    gemm_wmma<0><<<gH, blk, 0, stream>>>(h, Hh, wv, Hh, bv + (size_t)l * Hh,
                                         vb, Hh, MTOK, Hh, Hh);
    attn_kernel<<<(Bb * NHh * Tt) / 8, blk, 0, stream>>>(qb, kb, vb, cb);
    gemm_wmma<0><<<gH, blk, 0, stream>>>(cb, Hh, wo, Hh, bo + (size_t)l * Hh,
                                         tb, Hh, MTOK, Hh, Hh);
    add_ln_kernel<<<MTOK, blk, 0, stream>>>(h, tb, ln1g + (size_t)l * Hh,
                                            ln1b + (size_t)l * Hh, h);
    gemm_wmma<1><<<gFF, blk, 0, stream>>>(h, Hh, w1, FFf, b1 + (size_t)l * FFf,
                                          a1, FFf, MTOK, FFf, Hh);
    gemm_wmma<0><<<gH, blk, 0, stream>>>(a1, FFf, w2, Hh, b2 + (size_t)l * Hh,
                                         tb, Hh, MTOK, Hh, FFf);
    add_ln_kernel<<<MTOK, blk, 0, stream>>>(h, tb, ln2g + (size_t)l * Hh,
                                            ln2b + (size_t)l * Hh, h);
  }

  // Chunked vocab projection + streaming log-softmax NLL
  loss_init_kernel<<<(MTOK + 255) / 256, blk, 0, stream>>>(rmax, rsum, tlg);
  for (int n0 = 0; n0 < Vv; n0 += VC) {
    const int n = (Vv - n0 < VC) ? (Vv - n0) : VC;
    const dim3 gV((n + BN - 1) / BN, MTOK / BM);
    gemm_wmma<0><<<gV, blk, 0, stream>>>(h, Hh, Wc + n0, Vv, bc + n0,
                                         Lc, VC, MTOK, n, Hh);
    loss_chunk_kernel<<<MTOK, blk, 0, stream>>>(Lc, y, n0, n, rmax, rsum, tlg);
  }
  loss_final_kernel<<<1, blk, 0, stream>>>(rmax, rsum, tlg, (float*)d_out);
}
